// Transformer_Block_63161789055460
// MI455X (gfx1250) — compile-verified, matrix operand order checked by reading, not run on hardware
//
#include <hip/hip_runtime.h>

#define BATCH 4
#define NPTS  4096
#define KNN   16
#define DPT   64
#define DM    256
#define NTOT  (BATCH*NPTS)   // 16384 points total

typedef _Float16 v16h __attribute__((ext_vector_type(16)));
typedef _Float16 v8h  __attribute__((ext_vector_type(8)));
typedef float    v8f  __attribute__((ext_vector_type(8)));

union H16 { v16h v; v8h h[2]; };

// ---------------------------------------------------------------------------
// WMMA helpers (CDNA5: D = A(16x32 f16) x B(32x16 f16) + C(16x16 f32))
// ---------------------------------------------------------------------------
__device__ __forceinline__ v8f wmma16(v16h a, v16h b, v8f c) {
  return __builtin_amdgcn_wmma_f32_16x16x32_f16(
      /*neg_a=*/false, a, /*neg_b=*/false, b,
      /*c_mod=*/(short)0, c, /*reuse_a=*/false, /*reuse_b=*/false);
}

// A-fragment from row-major f16 matrix (global or LDS).
// Lane l: M = l&15 ; K halves = k0 + {0..7 | 8..15} and k0+16 + {0..7 | 8..15}
__device__ __forceinline__ v16h frag_a_h(const _Float16* base, int ld, int row0,
                                         int k0, int lane) {
  int m  = lane & 15;
  int ko = (lane & 16) ? 8 : 0;
  const _Float16* p = base + (size_t)(row0 + m) * ld + (k0 + ko);
  H16 u;
  u.h[0] = *(const v8h*)(p);
  u.h[1] = *(const v8h*)(p + 16);
  return u.v;
}

// A-fragment from row-major f32 matrix (convert to f16 on the fly).
__device__ __forceinline__ v16h frag_a_f(const float* base, int ld, int row0,
                                         int k0, int lane) {
  int m  = lane & 15;
  int ko = (lane & 16) ? 8 : 0;
  const float* p = base + (size_t)(row0 + m) * ld + (k0 + ko);
  float4 x0 = *(const float4*)(p);
  float4 x1 = *(const float4*)(p + 4);
  float4 y0 = *(const float4*)(p + 16);
  float4 y1 = *(const float4*)(p + 20);
  v16h r;
  r[0]=(_Float16)x0.x;  r[1]=(_Float16)x0.y;  r[2]=(_Float16)x0.z;  r[3]=(_Float16)x0.w;
  r[4]=(_Float16)x1.x;  r[5]=(_Float16)x1.y;  r[6]=(_Float16)x1.z;  r[7]=(_Float16)x1.w;
  r[8]=(_Float16)y0.x;  r[9]=(_Float16)y0.y;  r[10]=(_Float16)y0.z; r[11]=(_Float16)y0.w;
  r[12]=(_Float16)y1.x; r[13]=(_Float16)y1.y; r[14]=(_Float16)y1.z; r[15]=(_Float16)y1.w;
  return r;
}

// B-fragment from fragment-major pre-swizzled weight buffer: each block
// (ctile*ksteps + kstep) is 512 halves; lane l reads halves [l*16 .. l*16+15].
__device__ __forceinline__ v16h frag_b(const _Float16* fb, int blk, int lane) {
  const _Float16* p = fb + ((size_t)blk << 9) + ((size_t)lane << 4);
  H16 u;
  u.h[0] = *(const v8h*)(p);
  u.h[1] = *(const v8h*)(p + 8);
  return u.v;
}

// Store a 16x16 f32 accumulator tile into a row-major f16 buffer [16][DM].
__device__ __forceinline__ void store_tile_h(_Float16* buf, int ct, int lane,
                                             v8f acc, bool relu) {
  int col = ct * 16 + (lane & 15);
  int rb  = (lane & 16) ? 8 : 0;
#pragma unroll
  for (int r = 0; r < 8; ++r) {
    float v = acc[r];
    if (relu) v = fmaxf(v, 0.f);
    buf[(size_t)(rb + r) * DM + col] = (_Float16)v;
  }
}

// ---------------------------------------------------------------------------
// Kernel 0: weight -> WMMA B-fragment layout (f32 row-major [OUT][IN] -> f16)
// dst half index i = blk*512 + lane*16 + c*8 + j  maps to W[n][k]:
//   n = (blk/ksteps)*16 + lane%16 ; k = (blk%ksteps)*32 + c*16 + (lane<16?0:8) + j
// ---------------------------------------------------------------------------
__global__ void prep_frag(const float* __restrict__ W, _Float16* __restrict__ dst,
                          int IN, int total) {
  int i = blockIdx.x * 256 + threadIdx.x;
  if (i >= total) return;
  int j    = i & 7;
  int c    = (i >> 3) & 1;
  int lane = (i >> 4) & 31;
  int blk  = i >> 9;
  int ksteps = IN >> 5;
  int n = (blk / ksteps) * 16 + (lane & 15);
  int k = (blk % ksteps) * 32 + c * 16 + ((lane & 16) ? 8 : 0) + j;
  dst[i] = (_Float16)W[(size_t)n * IN + k];
}

// ---------------------------------------------------------------------------
// Kernel 1: kNN (K=16 incl. self) via LDS-tiled brute force + register
// insertion sort. Indices stored FLAT (b*N + j) to simplify later gathers.
// ---------------------------------------------------------------------------
__global__ __launch_bounds__(256) void knn_kernel(const float* __restrict__ xyz,
                                                  int* __restrict__ idxOut) {
  __shared__ float sx[256][3];
  int b  = blockIdx.x >> 4;
  int qt = blockIdx.x & 15;
  int qg = b * NPTS + qt * 256 + threadIdx.x;
  float qx = xyz[qg * 3 + 0], qy = xyz[qg * 3 + 1], qz = xyz[qg * 3 + 2];
  float dist[KNN];
  int   nn[KNN];
#pragma unroll
  for (int i = 0; i < KNN; ++i) { dist[i] = 3.4e38f; nn[i] = qg; }
  for (int tile = 0; tile < 16; ++tile) {
    int pg = b * NPTS + tile * 256 + threadIdx.x;
    sx[threadIdx.x][0] = xyz[pg * 3 + 0];
    sx[threadIdx.x][1] = xyz[pg * 3 + 1];
    sx[threadIdx.x][2] = xyz[pg * 3 + 2];
    __syncthreads();
    for (int jj = 0; jj < 256; ++jj) {
      float dx = qx - sx[jj][0], dy = qy - sx[jj][1], dz = qz - sx[jj][2];
      float d = dx * dx + dy * dy + dz * dz;
      if (d < dist[KNN - 1]) {
        dist[KNN - 1] = d;
        nn[KNN - 1]   = b * NPTS + tile * 256 + jj;
#pragma unroll
        for (int t = KNN - 1; t > 0; --t) {
          if (dist[t] < dist[t - 1]) {
            float td = dist[t]; dist[t] = dist[t - 1]; dist[t - 1] = td;
            int   ti = nn[t];   nn[t]   = nn[t - 1];   nn[t - 1]   = ti;
          }
        }
      }
    }
    __syncthreads();
  }
#pragma unroll
  for (int i = 0; i < KNN; ++i) idxOut[(size_t)qg * KNN + i] = nn[i];
}

// ---------------------------------------------------------------------------
// Kernel 2: inp = features @ W_pre^T + b_pre   (f32 in, f16 out) — WMMA
// One wave per 16x16 output tile. M=16384, K=64 (2 k-steps), N=256.
// ---------------------------------------------------------------------------
__global__ __launch_bounds__(256) void pre_gemm(const float* __restrict__ feat,
                                                const _Float16* __restrict__ fWpre,
                                                const float* __restrict__ b_pre,
                                                _Float16* __restrict__ inp) {
  int tid = threadIdx.x, wave = tid >> 5, lane = tid & 31;
  int tile = blockIdx.x * 8 + wave;       // 16384 tiles
  int mt = tile >> 4, ct = tile & 15;
  int row0 = mt * 16;
  v8f acc = {};
#pragma unroll
  for (int ks = 0; ks < 2; ++ks)
    acc = wmma16(frag_a_f(feat, DPT, row0, ks * 32, lane),
                 frag_b(fWpre, ct * 2 + ks, lane), acc);
  int col  = ct * 16 + (lane & 15);
  float bv = b_pre[col];
  int rb   = (lane & 16) ? 8 : 0;
#pragma unroll
  for (int r = 0; r < 8; ++r)
    inp[(size_t)(row0 + rb + r) * DM + col] = (_Float16)(acc[r] + bv);
}

// ---------------------------------------------------------------------------
// Kernel 3: {q,k,v} = LN(inp @ W^T)  — 16-row stripe per block, WMMA + fused LN
// blockIdx.y selects Q/K/V.  A-fragments hoisted (shared by both ctiles).
// ---------------------------------------------------------------------------
__global__ __launch_bounds__(256) void qkv_ln_kernel(
    const _Float16* __restrict__ inp,
    const _Float16* __restrict__ fWQ, const _Float16* __restrict__ fWK,
    const _Float16* __restrict__ fWV,
    const float* __restrict__ g, const float* __restrict__ bb,
    _Float16* __restrict__ qo, _Float16* __restrict__ ko,
    _Float16* __restrict__ vo) {
  __shared__ __align__(16) float stripe[16][DM];
  int tid = threadIdx.x, wave = tid >> 5, lane = tid & 31;
  int row0 = blockIdx.x * 16;
  const _Float16* WF = (blockIdx.y == 0) ? fWQ : (blockIdx.y == 1) ? fWK : fWV;
  _Float16* outp     = (blockIdx.y == 0) ? qo  : (blockIdx.y == 1) ? ko  : vo;

  v16h afrag[8];
#pragma unroll
  for (int ks = 0; ks < 8; ++ks)
    afrag[ks] = frag_a_h(inp, DM, row0, ks * 32, lane);

#pragma unroll
  for (int ci = 0; ci < 2; ++ci) {
    int ct = wave * 2 + ci;
    v8f acc = {};
#pragma unroll
    for (int ks = 0; ks < 8; ++ks)
      acc = wmma16(afrag[ks], frag_b(WF, ct * 8 + ks, lane), acc);
    int col = ct * 16 + (lane & 15);
    int rb  = (lane & 16) ? 8 : 0;
#pragma unroll
    for (int r = 0; r < 8; ++r) stripe[rb + r][col] = acc[r];
  }
  __syncthreads();

  // LayerNorm: 16 threads per row; shuffle-xor reduction stays inside wave32.
  int r = tid >> 4, sub = tid & 15;
  float sum = 0.f, ss = 0.f;
#pragma unroll
  for (int t = 0; t < 16; ++t) {
    float x = stripe[r][sub + 16 * t];
    sum += x; ss += x * x;
  }
#pragma unroll
  for (int m = 1; m < 16; m <<= 1) {
    sum += __shfl_xor(sum, m, 32);
    ss  += __shfl_xor(ss,  m, 32);
  }
  float mu  = sum * (1.0f / DM);
  float var = ss * (1.0f / DM) - mu * mu;
  float inv = rsqrtf(var + 1e-5f);
#pragma unroll
  for (int t = 0; t < 16; ++t) {
    int c = sub + 16 * t;
    outp[(size_t)(row0 + r) * DM + c] =
        (_Float16)((stripe[r][c] - mu) * inv * g[c] + bb[c]);
  }
}

// ---------------------------------------------------------------------------
// GEMM for the fused attention kernel:  D(16x256 f16, in LDS) =
//   op( A(16x256 f16, in LDS) @ Wfrag )   with A-fragments register-hoisted
//   and next-ctile B prefetch (global_prefetch_b8, L2-resident weights).
// ---------------------------------------------------------------------------
__device__ __forceinline__ void gemm_16x256(const _Float16* aBase,
                                            const _Float16* fW,
                                            _Float16* dBase, int lane,
                                            bool relu) {
  v16h afrag[8];
#pragma unroll
  for (int ks = 0; ks < 8; ++ks)
    afrag[ks] = frag_a_h(aBase, DM, 0, ks * 32, lane);

  for (int ct = 0; ct < 16; ++ct) {
    if (ct + 1 < 16)  // prefetch next ctile's fragment block (hits L2)
      __builtin_prefetch(fW + (((size_t)(ct + 1) * 8) << 9) + (lane << 4), 0, 3);
    v8f acc = {};
#pragma unroll
    for (int ks = 0; ks < 8; ++ks)
      acc = wmma16(afrag[ks], frag_b(fW, ct * 8 + ks, lane), acc);
    store_tile_h(dBase, ct, lane, acc, relu);
  }
}

// ---------------------------------------------------------------------------
// Kernel 4: fused per-point attention. One wave = one point; 8 points/block.
// LDS per point: A/B/POS f16 [16][256] (3x8KB) + small misc  (~202KB total).
// Per point: 3 GEMMs x 16 ctiles x 8 ksteps = 384 v_wmma.
// ---------------------------------------------------------------------------
__global__ __launch_bounds__(256) void attn_fused(
    const float* __restrict__ xyz, const int* __restrict__ idx,
    const float* __restrict__ Wp1,
    const _Float16* __restrict__ qf, const _Float16* __restrict__ kf,
    const _Float16* __restrict__ vf,
    const _Float16* __restrict__ fWp2, const _Float16* __restrict__ fWa1,
    const _Float16* __restrict__ fWa2,
    float* __restrict__ resOut) {
  __shared__ __align__(16) _Float16 bufA[8][16][DM];   // h -> t -> logits
  __shared__ __align__(16) _Float16 bufB[8][16][DM];   // u -> gathered v
  __shared__ __align__(16) _Float16 bufP[8][16][DM];   // pos
  __shared__ __align__(16) _Float16 sQ[8][DM];
  __shared__ int   sIdx[8][KNN];
  __shared__ float sRel[8][KNN][4];
  __shared__ float sWp1[DM * 3];

  const int tid = threadIdx.x, wave = tid >> 5, lane = tid & 31;
  const int point = blockIdx.x * 8 + wave;

  for (int i = tid; i < DM * 3; i += 256) sWp1[i] = Wp1[i];
  if (lane < KNN) {
    int nb = idx[(size_t)point * KNN + lane];
    sIdx[wave][lane]   = nb;
    sRel[wave][lane][0] = xyz[point * 3 + 0] - xyz[nb * 3 + 0];
    sRel[wave][lane][1] = xyz[point * 3 + 1] - xyz[nb * 3 + 1];
    sRel[wave][lane][2] = xyz[point * 3 + 2] - xyz[nb * 3 + 2];
  }
  *(v8h*)&sQ[wave][lane * 8] = *(const v8h*)(qf + (size_t)point * DM + lane * 8);
  __syncthreads();

  // phase 1: h = relu(rel @ Wp1^T)  -> bufA (tiny K=3, plain VALU)
  {
    int j  = lane & 15;
    int d0 = (lane & 16) ? 128 : 0;
    float rx = sRel[wave][j][0], ry = sRel[wave][j][1], rz = sRel[wave][j][2];
    for (int d = d0; d < d0 + 128; ++d) {
      float h = rx * sWp1[d * 3] + ry * sWp1[d * 3 + 1] + rz * sWp1[d * 3 + 2];
      bufA[wave][j][d] = (_Float16)fmaxf(h, 0.f);
    }
  }
  __syncthreads();

  // phase 2: pos = h @ Wp2^T  -> bufP   (128 WMMA)
  gemm_16x256(&bufA[wave][0][0], fWp2, &bufP[wave][0][0], lane, false);
  __syncthreads();

  // phase 3: t = q - k_gather + pos  -> bufA (h dead)
  for (int e = lane; e < 512; e += 32) {
    int j = e >> 5, d = (e & 31) * 8;
    int nb = sIdx[wave][j];
    v8h kk = *(const v8h*)(kf + (size_t)nb * DM + d);
    v8h qq = *(const v8h*)&sQ[wave][d];
    v8h pp = *(const v8h*)&bufP[wave][j][d];
    *(v8h*)&bufA[wave][j][d] = qq - kk + pp;
  }
  __syncthreads();

  // phase 4: u = relu(t @ Wa1^T) -> bufB   (128 WMMA)
  gemm_16x256(&bufA[wave][0][0], fWa1, &bufB[wave][0][0], lane, true);
  __syncthreads();

  // phase 5: logits = u @ Wa2^T -> bufA (t dead)   (128 WMMA)
  gemm_16x256(&bufB[wave][0][0], fWa2, &bufA[wave][0][0], lane, false);
  __syncthreads();

  // phase 5.5: gather v rows -> bufB (u dead)
  for (int e = lane; e < 512; e += 32) {
    int j = e >> 5, d = (e & 31) * 8;
    int nb = sIdx[wave][j];
    *(v8h*)&bufB[wave][j][d] = *(const v8h*)(vf + (size_t)nb * DM + d);
  }
  __syncthreads();

  // phase 6: softmax over the 16 neighbors per channel + weighted reduce
  for (int s = 0; s < 8; ++s) {
    int ch = lane + 32 * s;
    float mx = -1e30f;
#pragma unroll
    for (int j = 0; j < KNN; ++j) mx = fmaxf(mx, (float)bufA[wave][j][ch]);
    float den = 0.f, num = 0.f;
#pragma unroll
    for (int j = 0; j < KNN; ++j) {
      float w = __expf(((float)bufA[wave][j][ch] - mx) * 0.0625f); // /sqrt(256)
      den += w;
      num += w * ((float)bufB[wave][j][ch] + (float)bufP[wave][j][ch]);
    }
    resOut[(size_t)point * DM + ch] = num / den;
  }
}

// ---------------------------------------------------------------------------
// Kernel 5: out = LN(LN(res@Wproj^T)@Wpost^T + b_post) + features
// ---------------------------------------------------------------------------
__global__ __launch_bounds__(256) void final_kernel(
    const float* __restrict__ res, const _Float16* __restrict__ fWproj,
    const float* __restrict__ g_dm, const float* __restrict__ b_dm,
    const _Float16* __restrict__ fWpost, const float* __restrict__ b_post,
    const float* __restrict__ g_dp, const float* __restrict__ b_dp,
    const float* __restrict__ feat, float* __restrict__ out) {
  __shared__ __align__(16) float    stripe[16][DM];
  __shared__ __align__(16) _Float16 aBuf[16][DM];
  __shared__ __align__(16) float    stripe2[16][DPT];
  int tid = threadIdx.x, wave = tid >> 5, lane = tid & 31;
  int row0 = blockIdx.x * 16;

  // GEMM1: res @ Wproj^T  (A-fragments hoisted across the 2 ctiles)
  {
    v16h afrag[8];
#pragma unroll
    for (int ks = 0; ks < 8; ++ks)
      afrag[ks] = frag_a_f(res, DM, row0, ks * 32, lane);
#pragma unroll
    for (int ci = 0; ci < 2; ++ci) {
      int ct = wave * 2 + ci;
      v8f acc = {};
#pragma unroll
      for (int ks = 0; ks < 8; ++ks)
        acc = wmma16(afrag[ks], frag_b(fWproj, ct * 8 + ks, lane), acc);
      int col = ct * 16 + (lane & 15);
      int rb  = (lane & 16) ? 8 : 0;
#pragma unroll
      for (int r = 0; r < 8; ++r) stripe[rb + r][col] = acc[r];
    }
  }
  __syncthreads();

  // LN1 -> aBuf (f16 A for GEMM2)
  {
    int r = tid >> 4, sub = tid & 15;
    float sum = 0.f, ss = 0.f;
#pragma unroll
    for (int t = 0; t < 16; ++t) {
      float x = stripe[r][sub + 16 * t];
      sum += x; ss += x * x;
    }
#pragma unroll
    for (int m = 1; m < 16; m <<= 1) {
      sum += __shfl_xor(sum, m, 32);
      ss  += __shfl_xor(ss,  m, 32);
    }
    float mu = sum * (1.0f / DM), var = ss * (1.0f / DM) - mu * mu;
    float inv = rsqrtf(var + 1e-5f);
#pragma unroll
    for (int t = 0; t < 16; ++t) {
      int c = sub + 16 * t;
      aBuf[r][c] = (_Float16)((stripe[r][c] - mu) * inv * g_dm[c] + b_dm[c]);
    }
  }
  __syncthreads();

  // GEMM2: aBuf @ Wpost^T + b_post  (64 output channels: 4 ctiles, waves 0..3)
  if (wave < 4) {
    int ct = wave;
    v8f acc = {};
#pragma unroll
    for (int ks = 0; ks < 8; ++ks)
      acc = wmma16(frag_a_h(&aBuf[0][0], DM, 0, ks * 32, lane),
                   frag_b(fWpost, ct * 8 + ks, lane), acc);
    int col = ct * 16 + (lane & 15);
    int rb  = (lane & 16) ? 8 : 0;
    float bv = b_post[col];
#pragma unroll
    for (int r = 0; r < 8; ++r) stripe2[rb + r][col] = acc[r] + bv;
  }
  __syncthreads();

  // LN2 + residual -> out (f32 [16384][64])
  {
    int r = tid >> 4, sub = tid & 15;
    float sum = 0.f, ss = 0.f;
#pragma unroll
    for (int t = 0; t < 4; ++t) {
      float x = stripe2[r][sub + 16 * t];
      sum += x; ss += x * x;
    }
#pragma unroll
    for (int m = 1; m < 16; m <<= 1) {
      sum += __shfl_xor(sum, m, 32);
      ss  += __shfl_xor(ss,  m, 32);
    }
    float mu = sum * (1.0f / DPT), var = ss * (1.0f / DPT) - mu * mu;
    float inv = rsqrtf(var + 1e-5f);
#pragma unroll
    for (int t = 0; t < 4; ++t) {
      int c = sub + 16 * t;
      out[(size_t)(row0 + r) * DPT + c] =
          (stripe2[r][c] - mu) * inv * g_dp[c] + b_dp[c] +
          feat[(size_t)(row0 + r) * DPT + c];
    }
  }
}

// ---------------------------------------------------------------------------
extern "C" void kernel_launch(void* const* d_in, const int* in_sizes, int n_in,
                              void* d_out, int out_size, void* d_ws, size_t ws_size,
                              hipStream_t stream) {
  (void)in_sizes; (void)n_in; (void)out_size; (void)ws_size;
  const float* xyz    = (const float*)d_in[0];
  const float* feat   = (const float*)d_in[1];
  const float* W_pre  = (const float*)d_in[2];
  const float* b_pre  = (const float*)d_in[3];
  const float* W_post = (const float*)d_in[4];
  const float* b_post = (const float*)d_in[5];
  const float* Wp1    = (const float*)d_in[6];
  const float* Wp2    = (const float*)d_in[7];
  const float* Wa1    = (const float*)d_in[8];
  const float* Wa2    = (const float*)d_in[9];
  const float* WQ     = (const float*)d_in[10];
  const float* WK     = (const float*)d_in[11];
  const float* WV     = (const float*)d_in[12];
  const float* Wproj  = (const float*)d_in[13];
  const float* g_dm   = (const float*)d_in[14];
  const float* b_dm   = (const float*)d_in[15];
  const float* g_dp   = (const float*)d_in[16];
  const float* b_dp   = (const float*)d_in[17];
  float* outp = (float*)d_out;

  char* ws = (char*)d_ws;
  size_t off = 0;
  auto take = [&](size_t bytes) -> void* {
    void* p = ws + off;
    off += (bytes + 255) & ~(size_t)255;
    return p;
  };
  int*      wsIdx = (int*)      take((size_t)NTOT * KNN * sizeof(int));
  _Float16* wsInp = (_Float16*) take((size_t)NTOT * DM * 2);
  _Float16* wsQ   = (_Float16*) take((size_t)NTOT * DM * 2);
  _Float16* wsK   = (_Float16*) take((size_t)NTOT * DM * 2);
  _Float16* wsV   = (_Float16*) take((size_t)NTOT * DM * 2);
  float*    wsRes = (float*)    take((size_t)NTOT * DM * 4);
  _Float16* fPre  = (_Float16*) take((size_t)DM * DPT * 2);
  _Float16* fQ    = (_Float16*) take((size_t)DM * DM * 2);
  _Float16* fK    = (_Float16*) take((size_t)DM * DM * 2);
  _Float16* fV    = (_Float16*) take((size_t)DM * DM * 2);
  _Float16* fP2   = (_Float16*) take((size_t)DM * DM * 2);
  _Float16* fA1   = (_Float16*) take((size_t)DM * DM * 2);
  _Float16* fA2   = (_Float16*) take((size_t)DM * DM * 2);
  _Float16* fProj = (_Float16*) take((size_t)DM * DM * 2);
  _Float16* fPost = (_Float16*) take((size_t)DPT * DM * 2);

  // weight swizzle (all tiny; L2-resident afterwards)
  prep_frag<<<(DM * DPT + 255) / 256, 256, 0, stream>>>(W_pre, fPre, DPT, DM * DPT);
  prep_frag<<<(DM * DM + 255) / 256, 256, 0, stream>>>(WQ,    fQ,    DM,  DM * DM);
  prep_frag<<<(DM * DM + 255) / 256, 256, 0, stream>>>(WK,    fK,    DM,  DM * DM);
  prep_frag<<<(DM * DM + 255) / 256, 256, 0, stream>>>(WV,    fV,    DM,  DM * DM);
  prep_frag<<<(DM * DM + 255) / 256, 256, 0, stream>>>(Wp2,   fP2,   DM,  DM * DM);
  prep_frag<<<(DM * DM + 255) / 256, 256, 0, stream>>>(Wa1,   fA1,   DM,  DM * DM);
  prep_frag<<<(DM * DM + 255) / 256, 256, 0, stream>>>(Wa2,   fA2,   DM,  DM * DM);
  prep_frag<<<(DM * DM + 255) / 256, 256, 0, stream>>>(Wproj, fProj, DM,  DM * DM);
  prep_frag<<<(DPT * DM + 255) / 256, 256, 0, stream>>>(W_post, fPost, DM, DPT * DM);

  knn_kernel<<<BATCH * (NPTS / 256), 256, 0, stream>>>(xyz, wsIdx);
  pre_gemm<<<(NTOT / 16) * (DM / 16) / 8, 256, 0, stream>>>(feat, fPre, b_pre, wsInp);
  qkv_ln_kernel<<<dim3(NTOT / 16, 3, 1), 256, 0, stream>>>(
      wsInp, fQ, fK, fV, g_dm, b_dm, wsQ, wsK, wsV);
  attn_fused<<<NTOT / 8, 256, 0, stream>>>(
      xyz, wsIdx, Wp1, wsQ, wsK, wsV, fP2, fA1, fA2, wsRes);
  final_kernel<<<NTOT / 16, 256, 0, stream>>>(
      wsRes, fProj, g_dm, b_dm, fPost, b_post, g_dp, b_dp, feat, outp);
}
